// VLMMoE_48627619726052
// MI455X (gfx1250) — matrix operand order repaired and checked by reading, not compiled, not run on hardware
//
#include <hip/hip_runtime.h>

// ---------------------------------------------------------------------------
// MoE top-1 routed SwiGLU for MI455X (gfx1250, wave32, WMMA bf16).
// B=2,S=2048 -> T=4096 tokens; D=1024; F=4096; E=8.
// - Route first, compute only the top-1 expert per token (8x FLOP cut).
// - Grouped GEMMs with v_wmma_f32_16x16x32_bf16, fp32 accumulation.
// - Double-buffered LDS, one barrier per K-step; tile i+1 global loads overlap
//   WMMA on tile i.
// - Fragment-major B layout: fragments load as 2x ds_load_b128; B publish is
//   4x ds_store_b32 of v_cvt_pk_bf16_f32-packed K-pairs.
// - Down-GEMM A tiles (bf16 act) streamed with global_load_async_to_lds_b128
//   + s_wait_asynccnt (CDNA5 async copy path, overlaps WMMA).
// - M-tiles spread over gridDim.z for occupancy (early-exit on empty tiles).
// Weight traffic ~384MB fp32 -> ~16.5us HBM floor at 23.3 TB/s; per-expert
// weight blocks (<=64MB) fit the 192MB L2 so M-tile rereads are L2 hits.
// ---------------------------------------------------------------------------

typedef __attribute__((ext_vector_type(16))) __bf16 v16bf;
typedef __attribute__((ext_vector_type(2)))  __bf16 v2bf;
typedef __attribute__((ext_vector_type(8)))  float  v8f;
typedef __attribute__((ext_vector_type(2)))  float  v2f;

#define T_TOK 4096
#define D_DIM 1024
#define F_DIM 4096
#define E_NUM 8
#define MAXMT 64                 // ceil(T_TOK / 64) worst-case M tiles

#define LDA   40                 // A tile row stride (bf16): 64 rows x 32 K (+pad)
#define ABUF  (64 * LDA)         // elements per A buffer
#define LDBF  40                 // B tile (2n+h)-row stride (bf16): 16 K + pad
#define BBUF  (128 * LDBF)       // elements per B buffer (64 cols * 2 halves)

#define WMMA_BF16(A, B, C) \
  __builtin_amdgcn_wmma_f32_16x16x32_bf16(false, (A), false, (B), (short)0, (C), false, false)

__device__ __forceinline__ unsigned short f2bf(float f) {
  __bf16 h = (__bf16)f;
  return __builtin_bit_cast(unsigned short, h);
}
// Two floats -> packed bf16x2 word via a true vector conversion, so the
// backend selects a single two-source v_cvt_pk_bf16_f32.
__device__ __forceinline__ unsigned pack2bf(float lo, float hi) {
  v2f v = {lo, hi};
  v2bf r = __builtin_convertvector(v, v2bf);
  return __builtin_bit_cast(unsigned, r);
}

// CDNA5 async VMEM->LDS copy (16B per lane), tracked by ASYNCcnt.
__device__ __forceinline__ void async_load_b128(unsigned lds_off, const void* src) {
  asm volatile("global_load_async_to_lds_b128 %0, %1, off"
               :: "v"(lds_off), "v"(src) : "memory");
}
__device__ __forceinline__ void wait_asynccnt0() {
  asm volatile("s_wait_asynccnt 0" ::: "memory");
}

// A fragment: 16x32 bf16. Lane l: m = m0 + l%16, half h = l/16.
// VGPR v: K = h*16 + v*2  -> one 16-element run = 2x b128.
__device__ __forceinline__ void frag_a_load(const __bf16* sA, int m0, v16bf& a) {
  const int l = threadIdx.x & 31;
  const __bf16* base = sA + (m0 + (l & 15)) * LDA;
  const int h16 = (l >> 4) * 16;
#pragma unroll
  for (int v = 0; v < 8; ++v) {
    const int k = h16 + v * 2;
    a[2 * v]     = base[k];
    a[2 * v + 1] = base[k + 1];
  }
}

// B fragment: 32x16 bf16. Lane l: n = n0 + l%16, h = l/16; VGPR v holds
// K = h*16 + 2v,2v+1. Fragment-major LDS layout: element (k,n) stored at
// sB[(n*2 + (k>>4))*LDBF + (k&15)] -> each lane reads 16 consecutive bf16
// (32B, 16B-aligned) = 2x ds_load_b128.
__device__ __forceinline__ void frag_b_load(const __bf16* sB, int n0, v16bf& b) {
  const int l = threadIdx.x & 31;
  const __bf16* p = sB + ((n0 + (l & 15)) * 2 + (l >> 4)) * LDBF;
#pragma unroll
  for (int j = 0; j < 16; ++j) b[j] = p[j];
}

// ---------------------------------------------------------------------------
__global__ void moe_init_kernel(int* counts, int* cursor) {
  const int i = threadIdx.x;
  if (i < E_NUM) { counts[i] = 0; cursor[i] = 0; }
}

// One wave per token: 8 router dots over D=1024, softmax, first-max argmax.
__global__ __launch_bounds__(256)
void moe_router_kernel(const float* __restrict__ x, const float* __restrict__ rw,
                       const float* __restrict__ rb, float* __restrict__ logits,
                       float* __restrict__ probs, int* __restrict__ top1,
                       int* __restrict__ counts, float* __restrict__ lsesq) {
  const int lane = threadIdx.x & 31;
  const int t = blockIdx.x * 8 + (threadIdx.x >> 5);
  float acc[E_NUM];
#pragma unroll
  for (int e = 0; e < E_NUM; ++e) acc[e] = 0.f;
  const float* xt = x + (size_t)t * D_DIM;
  for (int d = lane; d < D_DIM; d += 32) {
    const float xv = xt[d];
#pragma unroll
    for (int e = 0; e < E_NUM; ++e) acc[e] += xv * rw[e * D_DIM + d];
  }
#pragma unroll
  for (int e = 0; e < E_NUM; ++e) {
    float v = acc[e];
#pragma unroll
    for (int off = 16; off > 0; off >>= 1) v += __shfl_xor(v, off, 32);
    acc[e] = v;
  }
  if (lane == 0) {
    float l[E_NUM];
    float m = -1e30f;
    int bi = 0;
#pragma unroll
    for (int e = 0; e < E_NUM; ++e) {
      l[e] = acc[e] + rb[e];
      if (l[e] > m) { m = l[e]; bi = e; }   // strict > => first max (jnp.argmax)
    }
    float s = 0.f;
    float p[E_NUM];
#pragma unroll
    for (int e = 0; e < E_NUM; ++e) { p[e] = __expf(l[e] - m); s += p[e]; }
    const float inv = 1.f / s;
#pragma unroll
    for (int e = 0; e < E_NUM; ++e) {
      logits[t * E_NUM + e] = l[e];
      probs[t * E_NUM + e]  = p[e] * inv;
    }
    top1[t] = bi;
    atomicAdd(&counts[bi], 1);               // integer atomics: deterministic
    const float lse = m + __logf(s);
    lsesq[t] = lse * lse;
  }
}

// Deterministic loss reductions + exclusive scan of counts -> offsets.
__global__ __launch_bounds__(256)
void moe_finalize_kernel(const float* __restrict__ probs, const float* __restrict__ lsesq,
                         const int* __restrict__ counts, int* __restrict__ offsets,
                         float* __restrict__ losses) {
  __shared__ float red[256];
  __shared__ float psum[E_NUM];
  const int tid = threadIdx.x;

  float z = 0.f;
  for (int t = tid; t < T_TOK; t += 256) z += lsesq[t];
  red[tid] = z; __syncthreads();
  for (int s = 128; s > 0; s >>= 1) { if (tid < s) red[tid] += red[tid + s]; __syncthreads(); }
  const float ztot = red[0];
  __syncthreads();

  for (int e = 0; e < E_NUM; ++e) {
    float p = 0.f;
    for (int t = tid; t < T_TOK; t += 256) p += probs[t * E_NUM + e];
    red[tid] = p; __syncthreads();
    for (int s = 128; s > 0; s >>= 1) { if (tid < s) red[tid] += red[tid + s]; __syncthreads(); }
    if (tid == 0) psum[e] = red[0];
    __syncthreads();
  }

  if (tid == 0) {
    int acc = 0;
    float lb = 0.f;
    offsets[0] = 0;
    for (int e = 0; e < E_NUM; ++e) {
      lb += ((float)counts[e] / (float)T_TOK) * psum[e];
      acc += counts[e];
      offsets[e + 1] = acc;
    }
    losses[0] = lb * (float)E_NUM;                    // load balancing loss
    losses[1] = (ztot / (float)T_TOK) * (float)E_NUM; // router z-loss
  }
}

__global__ void moe_scatter_kernel(const int* __restrict__ top1, const int* __restrict__ offsets,
                                   int* __restrict__ cursor, int* __restrict__ perm) {
  const int t = blockIdx.x * blockDim.x + threadIdx.x;
  if (t < T_TOK) {
    const int e = top1[t];
    const int pos = atomicAdd(&cursor[e], 1);
    perm[offsets[e] + pos] = t;
  }
}

// ---------------------------------------------------------------------------
// Grouped GEMM 1: act[r, f] = silu(x_r . Wg[:,f]) * (x_r . Wu[:,f]), bf16 out.
// grid = (F/64, E, MAXMT); block tile 64 tokens x 64 F-cols, 8 waves.
// ---------------------------------------------------------------------------
__global__ __launch_bounds__(256)
void moe_gateup_kernel(const float* __restrict__ x, const float* __restrict__ wg,
                       const float* __restrict__ wu, const int* __restrict__ perm,
                       const int* __restrict__ counts, const int* __restrict__ offsets,
                       unsigned short* __restrict__ act) {
  const int e = blockIdx.y;
  const int cnt = counts[e];
  const int mbase = blockIdx.z * 64;
  if (mbase >= cnt) return;
  const int offs = offsets[e];
  const int fbase = blockIdx.x * 64;
  const float* __restrict__ wgE = wg + (size_t)e * D_DIM * F_DIM;
  const float* __restrict__ wuE = wu + (size_t)e * D_DIM * F_DIM;

  __shared__ int sTok[64];
  __shared__ __align__(16) __bf16 sA[2 * ABUF];
  __shared__ __align__(16) __bf16 sBg[2 * BBUF];
  __shared__ __align__(16) __bf16 sBu[2 * BBUF];

  const int tid  = threadIdx.x;
  const int lane = tid & 31;
  const int wv   = tid >> 5;
  const int m0   = (wv & 3) * 16;
  const int n0   = (wv >> 2) * 32;

  // A loader: 64 rows x 32 K; thread -> row (tid>>2), 8 K at (tid&3)*8.
  const int arow = tid >> 2;
  const int aks  = (tid & 3) * 8;
  // B loader: 16 K-pairs x 64 cols; thread -> pair (tid>>4), 4 cols at (tid&15)*4.
  const int bp   = tid >> 4;            // K pair index, k0 = 2*bp
  const int bc4  = (tid & 15) * 4;      // col start
  const int bh   = bp >> 3;             // K half (k0>>4)
  const int bkk  = (2 * bp) & 15;       // even K within half

  if (tid < 64) {
    const int r = mbase + tid;
    sTok[tid] = (r < cnt) ? perm[offs + r] : -1;
  }
  __syncthreads();

  const int myTok = sTok[arow];
  const float4* aP = (myTok >= 0)
      ? (const float4*)(x + (size_t)myTok * D_DIM + aks) : (const float4*)0;
  const float* gP = wgE + (size_t)(2 * bp) * F_DIM + fbase + bc4;
  const float* uP = wuE + (size_t)(2 * bp) * F_DIM + fbase + bc4;

  v8f accg0 = {}, accg1 = {}, accu0 = {}, accu1 = {};

  // Prologue: tile 0 -> regs (g0/u0 = row k0, g1/u1 = row k0+1)
  float4 av0 = make_float4(0.f, 0.f, 0.f, 0.f), av1 = av0;
  if (aP) { av0 = aP[0]; av1 = aP[1]; }
  float4 g0 = ((const float4*)gP)[0], g1 = ((const float4*)(gP + F_DIM))[0];
  float4 u0 = ((const float4*)uP)[0], u1 = ((const float4*)(uP + F_DIM))[0];

  for (int i = 0; i < D_DIM / 32; ++i) {
    // 1) publish tile i: regs -> LDS buf[i&1]
    {
      __bf16* dA = sA + (i & 1) * ABUF + arow * LDA + aks;
      dA[0] = (__bf16)av0.x; dA[1] = (__bf16)av0.y; dA[2] = (__bf16)av0.z; dA[3] = (__bf16)av0.w;
      dA[4] = (__bf16)av1.x; dA[5] = (__bf16)av1.y; dA[6] = (__bf16)av1.z; dA[7] = (__bf16)av1.w;
      __bf16* bG = sBg + (i & 1) * BBUF;
      __bf16* bU = sBu + (i & 1) * BBUF;
      *(unsigned*)(bG + ((bc4 + 0) * 2 + bh) * LDBF + bkk) = pack2bf(g0.x, g1.x);
      *(unsigned*)(bG + ((bc4 + 1) * 2 + bh) * LDBF + bkk) = pack2bf(g0.y, g1.y);
      *(unsigned*)(bG + ((bc4 + 2) * 2 + bh) * LDBF + bkk) = pack2bf(g0.z, g1.z);
      *(unsigned*)(bG + ((bc4 + 3) * 2 + bh) * LDBF + bkk) = pack2bf(g0.w, g1.w);
      *(unsigned*)(bU + ((bc4 + 0) * 2 + bh) * LDBF + bkk) = pack2bf(u0.x, u1.x);
      *(unsigned*)(bU + ((bc4 + 1) * 2 + bh) * LDBF + bkk) = pack2bf(u0.y, u1.y);
      *(unsigned*)(bU + ((bc4 + 2) * 2 + bh) * LDBF + bkk) = pack2bf(u0.z, u1.z);
      *(unsigned*)(bU + ((bc4 + 3) * 2 + bh) * LDBF + bkk) = pack2bf(u0.w, u1.w);
    }
    __syncthreads();
    // 2) start global loads for tile i+1 (latency hidden behind WMMA)
    if (i + 1 < D_DIM / 32) {
      const int kt = (i + 1) * 32;
      if (aP) { av0 = aP[kt >> 2]; av1 = aP[(kt >> 2) + 1]; }
      const float* gQ = gP + (size_t)kt * F_DIM;
      const float* uQ = uP + (size_t)kt * F_DIM;
      g0 = ((const float4*)gQ)[0]; g1 = ((const float4*)(gQ + F_DIM))[0];
      u0 = ((const float4*)uQ)[0]; u1 = ((const float4*)(uQ + F_DIM))[0];
    }
    // 3) WMMA on tile i
    const __bf16* cA = sA  + (i & 1) * ABUF;
    const __bf16* cG = sBg + (i & 1) * BBUF;
    const __bf16* cU = sBu + (i & 1) * BBUF;
    v16bf a, b;
    frag_a_load(cA, m0, a);
    frag_b_load(cG, n0,      b); accg0 = WMMA_BF16(a, b, accg0);
    frag_b_load(cG, n0 + 16, b); accg1 = WMMA_BF16(a, b, accg1);
    frag_b_load(cU, n0,      b); accu0 = WMMA_BF16(a, b, accu0);
    frag_b_load(cU, n0 + 16, b); accu1 = WMMA_BF16(a, b, accu1);
  }

  // Epilogue: act = silu(gate) * up -> bf16. C layout: VGPR i -> M = i + 8*(l/16).
  const int ln = lane & 15;
  const int lh = lane >> 4;
#pragma unroll
  for (int i = 0; i < 8; ++i) {
    const int rl = mbase + m0 + lh * 8 + i;
    if (rl < cnt) {
      unsigned short* rowp = act + (size_t)(offs + rl) * F_DIM + fbase;
      const float gg0 = accg0[i];
      const float s0 = gg0 / (1.f + __expf(-gg0));
      rowp[n0 + ln] = f2bf(s0 * accu0[i]);
      const float gg1 = accg1[i];
      const float s1 = gg1 / (1.f + __expf(-gg1));
      rowp[n0 + 16 + ln] = f2bf(s1 * accu1[i]);
    }
  }
}

// ---------------------------------------------------------------------------
// Grouped GEMM 2: out[token, d] = act[r, :] . Wd[:, d], scattered via perm.
// grid = (D/64, E, MAXMT). A tiles (raw bf16) streamed via
// global_load_async_to_lds_b128 (ASYNCcnt), overlapping the WMMAs.
// ---------------------------------------------------------------------------
__global__ __launch_bounds__(256)
void moe_down_kernel(const unsigned short* __restrict__ act, const float* __restrict__ wd,
                     const int* __restrict__ perm, const int* __restrict__ counts,
                     const int* __restrict__ offsets, float* __restrict__ out) {
  const int e = blockIdx.y;
  const int cnt = counts[e];
  const int mbase = blockIdx.z * 64;
  if (mbase >= cnt) return;
  const int offs = offsets[e];
  const int dbase = blockIdx.x * 64;
  const float* __restrict__ wdE = wd + (size_t)e * F_DIM * D_DIM;

  __shared__ __align__(16) __bf16 sA[2 * ABUF];
  __shared__ __align__(16) __bf16 sB[2 * BBUF];

  const int tid  = threadIdx.x;
  const int lane = tid & 31;
  const int wv   = tid >> 5;
  const int m0   = (wv & 3) * 16;
  const int n0   = (wv >> 2) * 32;

  const int arow = tid >> 2;
  const int aks  = (tid & 3) * 8;
  const int bp   = tid >> 4;
  const int bc4  = (tid & 15) * 4;
  const int bh   = bp >> 3;
  const int bkk  = (2 * bp) & 15;

  const unsigned ldsA0 = (unsigned)(uintptr_t)(&sA[0]) + (unsigned)((arow * LDA + aks) * 2);
  const float* bP = wdE + (size_t)(2 * bp) * D_DIM + dbase + bc4;

  // Clamp padding rows into range; their outputs are never stored.
  const int rlc = (mbase + arow < cnt) ? (mbase + arow) : (cnt - 1);
  const unsigned short* aS = act + (size_t)(offs + rlc) * F_DIM + aks;

  v8f acc0 = {}, acc1 = {};

  // Prologue: async A(0) -> bufA0; B(0) -> regs (b0 = row k0, b1 = row k0+1)
  async_load_b128(ldsA0, aS);
  float4 b0 = ((const float4*)bP)[0], b1 = ((const float4*)(bP + D_DIM))[0];

  for (int i = 0; i < F_DIM / 32; ++i) {
    // 1) publish B tile i
    {
      __bf16* bB = sB + (i & 1) * BBUF;
      *(unsigned*)(bB + ((bc4 + 0) * 2 + bh) * LDBF + bkk) = pack2bf(b0.x, b1.x);
      *(unsigned*)(bB + ((bc4 + 1) * 2 + bh) * LDBF + bkk) = pack2bf(b0.y, b1.y);
      *(unsigned*)(bB + ((bc4 + 2) * 2 + bh) * LDBF + bkk) = pack2bf(b0.z, b1.z);
      *(unsigned*)(bB + ((bc4 + 3) * 2 + bh) * LDBF + bkk) = pack2bf(b0.w, b1.w);
    }
    wait_asynccnt0();              // A tile i landed in LDS
    __syncthreads();
    // 2) kick off tile i+1 (async A + B regs), hidden behind WMMA
    if (i + 1 < F_DIM / 32) {
      const int kt = (i + 1) * 32;
      async_load_b128(ldsA0 + (unsigned)(((i + 1) & 1) * ABUF * 2), aS + kt);
      const float* bQ = bP + (size_t)kt * D_DIM;
      b0 = ((const float4*)bQ)[0]; b1 = ((const float4*)(bQ + D_DIM))[0];
    }
    // 3) WMMA on tile i
    const __bf16* cA = sA + (i & 1) * ABUF;
    const __bf16* cB = sB + (i & 1) * BBUF;
    v16bf a, b;
    frag_a_load(cA, m0, a);
    frag_b_load(cB, n0,      b); acc0 = WMMA_BF16(a, b, acc0);
    frag_b_load(cB, n0 + 16, b); acc1 = WMMA_BF16(a, b, acc1);
  }

  const int ln = lane & 15;
  const int lh = lane >> 4;
#pragma unroll
  for (int i = 0; i < 8; ++i) {
    const int rl = mbase + m0 + lh * 8 + i;
    if (rl < cnt) {
      const int t = perm[offs + rl];
      float* rowp = out + (size_t)t * D_DIM + dbase;
      rowp[n0 + ln]      = acc0[i];
      rowp[n0 + 16 + ln] = acc1[i];
    }
  }
}

// ---------------------------------------------------------------------------
extern "C" void kernel_launch(void* const* d_in, const int* in_sizes, int n_in,
                              void* d_out, int out_size, void* d_ws, size_t ws_size,
                              hipStream_t stream) {
  (void)in_sizes; (void)n_in; (void)out_size; (void)ws_size;
  const float* x  = (const float*)d_in[0];
  const float* rw = (const float*)d_in[1];
  const float* rb = (const float*)d_in[2];
  const float* wg = (const float*)d_in[3];
  const float* wu = (const float*)d_in[4];
  const float* wd = (const float*)d_in[5];
  float* out = (float*)d_out;

  // d_out layout (reference return order): expert_outputs [T,D], probs [T,E],
  // logits [T,E], lb_loss, z_loss.
  float* probs  = out + (size_t)T_TOK * D_DIM;          // 4194304
  float* logits = probs + (size_t)T_TOK * E_NUM;        // +32768
  float* losses = logits + (size_t)T_TOK * E_NUM;       // +32768

  // Workspace layout
  char* ws = (char*)d_ws;
  int*   top1    = (int*)ws;                            // T ints
  int*   counts  = (int*)(ws + 16384);                  // E ints
  int*   offsets = (int*)(ws + 16384 + 64);             // E+1 ints
  int*   cursor  = (int*)(ws + 16384 + 128);            // E ints
  int*   perm    = (int*)(ws + 16384 + 256);            // T ints
  float* lsesq   = (float*)(ws + 16384 + 256 + 16384);  // T floats
  unsigned short* act = (unsigned short*)(ws + 65536);  // T*F bf16 = 32 MB

  moe_init_kernel<<<dim3(1), dim3(32), 0, stream>>>(counts, cursor);
  moe_router_kernel<<<dim3(T_TOK / 8), dim3(256), 0, stream>>>(
      x, rw, rb, logits, probs, top1, counts, lsesq);
  moe_finalize_kernel<<<dim3(1), dim3(256), 0, stream>>>(
      probs, lsesq, counts, offsets, losses);
  moe_scatter_kernel<<<dim3(T_TOK / 256), dim3(256), 0, stream>>>(
      top1, offsets, cursor, perm);
  moe_gateup_kernel<<<dim3(F_DIM / 64, E_NUM, MAXMT), dim3(256), 0, stream>>>(
      x, wg, wu, perm, counts, offsets, act);
  moe_down_kernel<<<dim3(D_DIM / 64, E_NUM, MAXMT), dim3(256), 0, stream>>>(
      act, wd, perm, counts, offsets, out);
}
